// FullAttnModule_49950469653119
// MI455X (gfx1250) — matrix operand order repaired and checked by reading, not compiled, not run on hardware
//
#include <hip/hip_runtime.h>

// ---------------------------------------------------------------------------
// Types for CDNA5 WMMA (wave32): A/B = 16 bf16 per lane, C/D = 8 f32 per lane
// ---------------------------------------------------------------------------
typedef __attribute__((ext_vector_type(16))) __bf16 bf16x16;
typedef __attribute__((ext_vector_type(8)))  float  f32x8;

union FragAB {
    bf16x16 v;
    uint4   q[2];
    unsigned short u[16];
};

__device__ __forceinline__ unsigned short f32_to_bf16(float f) {
    unsigned int u = __float_as_uint(f);
    u += 0x7FFFu + ((u >> 16) & 1u);   // round-to-nearest-even
    return (unsigned short)(u >> 16);
}

__device__ __forceinline__ f32x8 wmma_bf16(const FragAB& a, const FragAB& b, f32x8 c) {
    return __builtin_amdgcn_wmma_f32_16x16x32_bf16(
        /*neg_a=*/false, a.v, /*neg_b=*/false, b.v,
        /*c_mod=*/(short)0, c, /*reuse_a=*/false, /*reuse_b=*/false);
}

// 16-byte async copy global -> LDS (ASYNCcnt-tracked, no VGPR staging)
__device__ __forceinline__ void async_cp16(unsigned lds_off, const void* gptr) {
    asm volatile("global_load_async_to_lds_b128 %0, %1, off"
                 :: "v"(lds_off), "v"(gptr) : "memory");
}
__device__ __forceinline__ unsigned lds_addr(const void* p) {
    return (unsigned)(unsigned long long)p;   // low 32 bits of flat LDS addr = LDS offset
}

// ---------------------------------------------------------------------------
// fp32 -> bf16 conversion (grid-stride)
// ---------------------------------------------------------------------------
__global__ __launch_bounds__(256) void cvt_f32_bf16_kernel(
    const float* __restrict__ in, unsigned short* __restrict__ out, int n) {
    int i = blockIdx.x * blockDim.x + threadIdx.x;
    int stride = gridDim.x * blockDim.x;
    for (; i < n; i += stride) out[i] = f32_to_bf16(in[i]);
}

// ---------------------------------------------------------------------------
// Fused QKV projection: Y = X @ W^T  (M=4096, N=1024, K=1024, bf16 in/out)
// blockIdx.z selects {Q,K,V}. Q,K stored [b,h,n,64]; V stored transposed
// [b,h,64,n]. Block = 128 threads = 4 waves 2x2, wave tile 32x32, block 64x64.
// A/B tiles double-buffered in LDS via global_load_async_to_lds_b128.
// ---------------------------------------------------------------------------
__global__ __launch_bounds__(128) void gemm_qkv_kernel(
    const unsigned short* __restrict__ Xq, const unsigned short* __restrict__ Xk,
    const unsigned short* __restrict__ Xv,
    const unsigned short* __restrict__ Wq, const unsigned short* __restrict__ Wk,
    const unsigned short* __restrict__ Wv,
    unsigned short* __restrict__ Qh, unsigned short* __restrict__ Kh,
    unsigned short* __restrict__ Vt) {
    __shared__ __align__(16) unsigned short sA[2][64][40];   // 64 x 32 bf16, pad->40
    __shared__ __align__(16) unsigned short sB[2][64][40];

    const int mode = blockIdx.z;
    const unsigned short* X  = (mode == 0) ? Xq : (mode == 1) ? Xk : Xv;
    const unsigned short* W  = (mode == 0) ? Wq : (mode == 1) ? Wk : Wv;
    unsigned short*      Out = (mode == 0) ? Qh : (mode == 1) ? Kh : Vt;

    const int tid  = threadIdx.x;
    const int lane = tid & 31;
    const int wave = tid >> 5;
    const int half = lane >> 4;
    const int lrow = lane & 15;
    const int rtb = blockIdx.y * 64;                 // block M tile
    const int ctb = blockIdx.x * 64;                 // block N tile
    const int mw = (wave & 1) * 32;                  // wave offsets inside block tile
    const int nw = (wave >> 1) * 32;

    // stage A(64x32) + B(64x32) for K-chunk kk into buffer `buf`
    auto stage = [&](int buf, int kk) {
#pragma unroll
        for (int j = 0; j < 2; ++j) {
            int c   = tid + j * 128;                 // 0..255 : 16B chunk id
            int row = c >> 2, col = (c & 3) * 8;
            async_cp16(lds_addr(&sA[buf][row][col]),
                       X + (size_t)(rtb + row) * 1024 + kk + col);
            async_cp16(lds_addr(&sB[buf][row][col]),
                       W + (size_t)(ctb + row) * 1024 + kk + col);
        }
    };

    const f32x8 z = {0.f,0.f,0.f,0.f,0.f,0.f,0.f,0.f};
    f32x8 acc[2][2] = {{z, z}, {z, z}};

    stage(0, 0);
    for (int k0 = 0; k0 < 1024; k0 += 32) {
        const int cur = (k0 >> 5) & 1;
        if (k0 + 32 < 1024) {
            stage(cur ^ 1, k0 + 32);                       // overlap next slab with compute
            asm volatile("s_wait_asynccnt 0x4" ::: "memory");   // in-order: cur slab landed
        } else {
            asm volatile("s_wait_asynccnt 0x0" ::: "memory");
        }
        __syncthreads();

        FragAB a[2], b[2];
#pragma unroll
        for (int mi = 0; mi < 2; ++mi) {
            // A layout: row = lane&15; K = 8*half+e (e<8), 16+8*half+e (e>=8)
            const unsigned short* p = &sA[cur][mw + mi * 16 + lrow][8 * half];
            a[mi].q[0] = *(const uint4*)p;
            a[mi].q[1] = *(const uint4*)(p + 16);
        }
#pragma unroll
        for (int ni = 0; ni < 2; ++ni) {
            // B(k,n) = W[n][k]: N = lane&15; K = 8*half+e (e<8), 16+8*half+e (e>=8)
            const unsigned short* p = &sB[cur][nw + ni * 16 + lrow][8 * half];
            b[ni].q[0] = *(const uint4*)p;
            b[ni].q[1] = *(const uint4*)(p + 16);
        }
#pragma unroll
        for (int mi = 0; mi < 2; ++mi)
#pragma unroll
            for (int ni = 0; ni < 2; ++ni)
                acc[mi][ni] = wmma_bf16(a[mi], b[ni], acc[mi][ni]);

        asm volatile("s_wait_dscnt 0x0" ::: "memory");
        __syncthreads();                                   // cur buffer free for overwrite
    }

    // Store head-split (D layout: M = r + 8*half, N = lane&15)
#pragma unroll
    for (int mi = 0; mi < 2; ++mi)
#pragma unroll
        for (int ni = 0; ni < 2; ++ni)
#pragma unroll
            for (int r = 0; r < 8; ++r) {
                int row = rtb + mw + mi * 16 + half * 8 + r;  // token index (b*2048+n)
                int col = ctb + nw + ni * 16 + lrow;          // feature j = h*64+c
                int bb = row >> 11, nn = row & 2047;
                int hh = col >> 6,  cc = col & 63;
                size_t addr = (mode == 2)
                    ? (((size_t)(bb * 16 + hh) * 64 + cc) * 2048 + nn)     // V^T [b,h,c,n]
                    : (((size_t)(bb * 16 + hh) * 2048 + nn) * 64 + cc);    // [b,h,n,c]
                Out[addr] = f32_to_bf16(acc[mi][ni][r]);
            }
}

// ---------------------------------------------------------------------------
// Flash attention: one block = 64 q-rows of one (b,h); 4 waves x 16 rows.
// K and V^T slabs (64x64 bf16 each) double-buffered in LDS via async copies —
// all 4 waves share them (4x L2-traffic cut vs direct loads).
// Per 64-key step: 8 QK^T WMMAs + bias/scale + online softmax + 8 PV WMMAs.
// ---------------------------------------------------------------------------
__global__ __launch_bounds__(128) void flash_attn_kernel(
    const unsigned short* __restrict__ Qh, const unsigned short* __restrict__ Kh,
    const unsigned short* __restrict__ Vt, const float* __restrict__ Bias,
    unsigned short* __restrict__ AO) {
    __shared__ __align__(16) unsigned short sK[2][64][72];  // keys x c, pad->72
    __shared__ __align__(16) unsigned short sV[2][64][72];  // c x keys, pad->72
    __shared__ __align__(16) unsigned short sP[4][16][72];  // per-wave P transpose tile

    const int tid  = threadIdx.x;
    const int lane = tid & 31;
    const int wave = tid >> 5;
    const int half = lane >> 4;
    const int lrow = lane & 15;

    const int bh   = blockIdx.y;        // b*16 + h
    const int bidx = bh >> 4;
    const int hidx = bh & 15;
    const int q0   = blockIdx.x * 64 + wave * 16;

    const unsigned short* Qhead = Qh + (size_t)bh * 2048 * 64;
    const unsigned short* Khead = Kh + (size_t)bh * 2048 * 64;
    const unsigned short* Vhead = Vt + (size_t)bh * 64 * 2048;
    const float*          Bb    = Bias + (size_t)bidx * 2048 * 2048;

    // stage K slab [64 keys][64 c] and V^T slab [64 c][64 k] for key base kb
    auto stage = [&](int buf, int kb) {
#pragma unroll
        for (int j = 0; j < 4; ++j) {
            int c   = tid + j * 128;                 // 0..511 : 16B chunk id
            int row = c >> 3, col = (c & 7) * 8;
            async_cp16(lds_addr(&sK[buf][row][col]),
                       Khead + (size_t)(kb + row) * 64 + col);
            async_cp16(lds_addr(&sV[buf][row][col]),
                       Vhead + (size_t)row * 2048 + kb + col);
        }
    };

    // Q fragments (head-dim chunks 0 and 32) — loaded once, reused all k steps
    FragAB qf[2];
#pragma unroll
    for (int cc = 0; cc < 2; ++cc) {
        const unsigned short* p = Qhead + (size_t)(q0 + lrow) * 64 + cc * 32 + 8 * half;
        qf[cc].q[0] = *(const uint4*)p;
        qf[cc].q[1] = *(const uint4*)(p + 16);
    }

    const f32x8 z = {0.f,0.f,0.f,0.f,0.f,0.f,0.f,0.f};
    f32x8 O[4] = {z, z, z, z};
    float m[8], l[8];
#pragma unroll
    for (int r = 0; r < 8; ++r) { m[r] = -3.0e38f; l[r] = 0.f; }

    const float SCALE = 0.125f;  // HEAD_DIM^-0.5

    stage(0, 0);
    for (int kb = 0; kb < 2048; kb += 64) {
        const int cur = (kb >> 6) & 1;
        if (kb + 64 < 2048) {
            stage(cur ^ 1, kb + 64);
            asm volatile("s_wait_asynccnt 0x8" ::: "memory");   // cur slabs landed
        } else {
            asm volatile("s_wait_asynccnt 0x0" ::: "memory");
        }
        __syncthreads();

        // ---- S = Q K^T over this 64-key slab (4 subtiles of 16 keys) ----
        f32x8 S[4] = {z, z, z, z};
#pragma unroll
        for (int ks = 0; ks < 4; ++ks)
#pragma unroll
            for (int cc = 0; cc < 2; ++cc) {
                FragAB kf;  // B(c,k)=K[k][c]: table-order c per lane
                const unsigned short* p = &sK[cur][ks * 16 + lrow][cc * 32 + 8 * half];
                kf.q[0] = *(const uint4*)p;
                kf.q[1] = *(const uint4*)(p + 16);
                S[ks] = wmma_bf16(qf[cc], kf, S[ks]);
            }

        // ---- scale + bias (streamed once, never materialized) ----
#pragma unroll
        for (int ks = 0; ks < 4; ++ks)
#pragma unroll
            for (int r = 0; r < 8; ++r) {
                int qq = q0 + half * 8 + r;
                int kk = kb + ks * 16 + lrow;
                S[ks][r] = S[ks][r] * SCALE + Bb[(size_t)qq * 2048 + kk];
            }

        // ---- online softmax (row reductions across the 16-lane N group) ----
        float fac[8];
#pragma unroll
        for (int r = 0; r < 8; ++r) {
            float t = fmaxf(fmaxf(S[0][r], S[1][r]), fmaxf(S[2][r], S[3][r]));
#pragma unroll
            for (int d = 1; d < 16; d <<= 1) t = fmaxf(t, __shfl_xor(t, d, 32));
            float mn = fmaxf(m[r], t);
            fac[r] = __expf(m[r] - mn);
            m[r] = mn;
        }
#pragma unroll
        for (int r = 0; r < 8; ++r) {
            float s = 0.f;
#pragma unroll
            for (int ks = 0; ks < 4; ++ks) {
                float p = __expf(S[ks][r] - m[r]);
                S[ks][r] = p;
                s += p;
            }
#pragma unroll
            for (int d = 1; d < 16; d <<= 1) s += __shfl_xor(s, d, 32);
            l[r] = l[r] * fac[r] + s;
        }
#pragma unroll
        for (int cs = 0; cs < 4; ++cs)
#pragma unroll
            for (int r = 0; r < 8; ++r) O[cs][r] *= fac[r];

        // ---- P: D layout -> A layout via per-wave LDS tile (bf16) ----
#pragma unroll
        for (int ks = 0; ks < 4; ++ks)
#pragma unroll
            for (int r = 0; r < 8; ++r)
                sP[wave][half * 8 + r][ks * 16 + lrow] = f32_to_bf16(S[ks][r]);
        __builtin_amdgcn_wave_barrier();   // same-wave DS ops are in order; fence compiler
        __asm__ volatile("" ::: "memory");

        // ---- O += P V (2 K-chunks of 32 keys x 4 c-subtiles) ----
#pragma unroll
        for (int kc = 0; kc < 2; ++kc) {
            FragAB pf;
            const unsigned short* pp = &sP[wave][lrow][kc * 32 + 8 * half];
            pf.q[0] = *(const uint4*)pp;
            pf.q[1] = *(const uint4*)(pp + 16);
#pragma unroll
            for (int cs = 0; cs < 4; ++cs) {
                FragAB vf;  // B(k,c)=V^T[c][k]: table-order k per lane
                const unsigned short* vp = &sV[cur][cs * 16 + lrow][kc * 32 + 8 * half];
                vf.q[0] = *(const uint4*)vp;
                vf.q[1] = *(const uint4*)(vp + 16);
                O[cs] = wmma_bf16(pf, vf, O[cs]);
            }
        }

        asm volatile("s_wait_dscnt 0x0" ::: "memory");
        __syncthreads();                    // cur slabs free for overwrite
    }

    // ---- normalize and store to [b, n, h*64+c] bf16 (ready for Wo GEMM) ----
#pragma unroll
    for (int r = 0; r < 8; ++r) l[r] = 1.0f / l[r];
#pragma unroll
    for (int cs = 0; cs < 4; ++cs)
#pragma unroll
        for (int r = 0; r < 8; ++r) {
            int qq = q0 + half * 8 + r;
            int cc = cs * 16 + lrow;
            AO[((size_t)(bidx * 2048 + qq)) * 1024 + hidx * 64 + cc] =
                f32_to_bf16(O[cs][r] * l[r]);
        }
}

// ---------------------------------------------------------------------------
// Output projection: out = AO @ Wo^T + bo  (fp32 store to d_out)
// Same LDS-staged, async double-buffered structure as the QKV GEMM.
// ---------------------------------------------------------------------------
__global__ __launch_bounds__(128) void gemm_out_kernel(
    const unsigned short* __restrict__ X, const unsigned short* __restrict__ W,
    const float* __restrict__ bo, float* __restrict__ Out) {
    __shared__ __align__(16) unsigned short sA[2][64][40];
    __shared__ __align__(16) unsigned short sB[2][64][40];

    const int tid  = threadIdx.x;
    const int lane = tid & 31;
    const int wave = tid >> 5;
    const int half = lane >> 4;
    const int lrow = lane & 15;
    const int rtb = blockIdx.y * 64;
    const int ctb = blockIdx.x * 64;
    const int mw = (wave & 1) * 32;
    const int nw = (wave >> 1) * 32;

    auto stage = [&](int buf, int kk) {
#pragma unroll
        for (int j = 0; j < 2; ++j) {
            int c   = tid + j * 128;
            int row = c >> 2, col = (c & 3) * 8;
            async_cp16(lds_addr(&sA[buf][row][col]),
                       X + (size_t)(rtb + row) * 1024 + kk + col);
            async_cp16(lds_addr(&sB[buf][row][col]),
                       W + (size_t)(ctb + row) * 1024 + kk + col);
        }
    };

    const f32x8 z = {0.f,0.f,0.f,0.f,0.f,0.f,0.f,0.f};
    f32x8 acc[2][2] = {{z, z}, {z, z}};

    stage(0, 0);
    for (int k0 = 0; k0 < 1024; k0 += 32) {
        const int cur = (k0 >> 5) & 1;
        if (k0 + 32 < 1024) {
            stage(cur ^ 1, k0 + 32);
            asm volatile("s_wait_asynccnt 0x4" ::: "memory");
        } else {
            asm volatile("s_wait_asynccnt 0x0" ::: "memory");
        }
        __syncthreads();

        FragAB a[2], b[2];
#pragma unroll
        for (int mi = 0; mi < 2; ++mi) {
            const unsigned short* p = &sA[cur][mw + mi * 16 + lrow][8 * half];
            a[mi].q[0] = *(const uint4*)p;
            a[mi].q[1] = *(const uint4*)(p + 16);
        }
#pragma unroll
        for (int ni = 0; ni < 2; ++ni) {
            const unsigned short* p = &sB[cur][nw + ni * 16 + lrow][8 * half];
            b[ni].q[0] = *(const uint4*)p;
            b[ni].q[1] = *(const uint4*)(p + 16);
        }
#pragma unroll
        for (int mi = 0; mi < 2; ++mi)
#pragma unroll
            for (int ni = 0; ni < 2; ++ni)
                acc[mi][ni] = wmma_bf16(a[mi], b[ni], acc[mi][ni]);

        asm volatile("s_wait_dscnt 0x0" ::: "memory");
        __syncthreads();
    }

#pragma unroll
    for (int mi = 0; mi < 2; ++mi)
#pragma unroll
        for (int ni = 0; ni < 2; ++ni)
#pragma unroll
            for (int r = 0; r < 8; ++r) {
                int row = rtb + mw + mi * 16 + half * 8 + r;
                int col = ctb + nw + ni * 16 + lrow;
                Out[(size_t)row * 1024 + col] = acc[mi][ni][r] + bo[col];
            }
}

// ---------------------------------------------------------------------------
// Launch
// ---------------------------------------------------------------------------
extern "C" void kernel_launch(void* const* d_in, const int* in_sizes, int n_in,
                              void* d_out, int out_size, void* d_ws, size_t ws_size,
                              hipStream_t stream) {
    const float* query    = (const float*)d_in[0];
    const float* key      = (const float*)d_in[1];
    const float* value    = (const float*)d_in[2];
    const float* att_bias = (const float*)d_in[3];
    const float* Wq       = (const float*)d_in[4];
    const float* Wk       = (const float*)d_in[5];
    const float* Wv       = (const float*)d_in[6];
    const float* Wo       = (const float*)d_in[7];
    const float* bo       = (const float*)d_in[8];

    const size_t XE = (size_t)2 * 2048 * 1024;  // 4,194,304 elements
    const size_t WE = (size_t)1024 * 1024;      // 1,048,576 elements

    unsigned short* ws  = (unsigned short*)d_ws;
    unsigned short* Xqb = ws; ws += XE;
    unsigned short* Xkb = ws; ws += XE;
    unsigned short* Xvb = ws; ws += XE;
    unsigned short* Wqb = ws; ws += WE;
    unsigned short* Wkb = ws; ws += WE;
    unsigned short* Wvb = ws; ws += WE;
    unsigned short* Wob = ws; ws += WE;
    unsigned short* Qhb = ws; ws += XE;   // [b,h,n,64]
    unsigned short* Khb = ws; ws += XE;   // [b,h,n,64]
    unsigned short* Vtb = ws; ws += XE;   // [b,h,64,n]
    unsigned short* AOb = ws; ws += XE;   // [b,n,1024]  (total ws use = 64 MB)

    cvt_f32_bf16_kernel<<<dim3(1024), dim3(256), 0, stream>>>(query, Xqb, (int)XE);
    cvt_f32_bf16_kernel<<<dim3(1024), dim3(256), 0, stream>>>(key,   Xkb, (int)XE);
    cvt_f32_bf16_kernel<<<dim3(1024), dim3(256), 0, stream>>>(value, Xvb, (int)XE);
    cvt_f32_bf16_kernel<<<dim3(512),  dim3(256), 0, stream>>>(Wq, Wqb, (int)WE);
    cvt_f32_bf16_kernel<<<dim3(512),  dim3(256), 0, stream>>>(Wk, Wkb, (int)WE);
    cvt_f32_bf16_kernel<<<dim3(512),  dim3(256), 0, stream>>>(Wv, Wvb, (int)WE);
    cvt_f32_bf16_kernel<<<dim3(512),  dim3(256), 0, stream>>>(Wo, Wob, (int)WE);

    gemm_qkv_kernel<<<dim3(16, 64, 3), dim3(128), 0, stream>>>(
        Xqb, Xkb, Xvb, Wqb, Wkb, Wvb, Qhb, Khb, Vtb);

    flash_attn_kernel<<<dim3(32, 32), dim3(128), 0, stream>>>(
        Qhb, Khb, Vtb, att_bias, AOb);

    gemm_out_kernel<<<dim3(16, 64), dim3(128), 0, stream>>>(
        AOb, Wob, bo, (float*)d_out);
}
